// NCELoss_6090263625931
// MI455X (gfx1250) — matrix operand order repaired and checked by reading, not compiled, not run on hardware
//
#include <hip/hip_runtime.h>

typedef __attribute__((ext_vector_type(16))) _Float16 v16h;
typedef __attribute__((ext_vector_type(8)))  _Float16 v8h;
typedef __attribute__((ext_vector_type(4)))  _Float16 v4h;
typedef __attribute__((ext_vector_type(8)))  float    v8f;
typedef __attribute__((ext_vector_type(4)))  float    v4f;

#define N_ROWS 8192
#define DIM    128
#define CHUNKS 8
#define JC     (N_ROWS / CHUNKS)   // 1024 columns per chunk

union AFrag { v16h v; v8h h[2]; };

// ---------------------------------------------------------------------------
// Kernel 1: L2-normalize rows of f1 and f2, emit f16 copies into workspace.
// One wave per row: 32 lanes x float4 = 128 elements.
// ---------------------------------------------------------------------------
__global__ void __launch_bounds__(256)
nce_normalize_kernel(const float* __restrict__ f1,
                     const float* __restrict__ f2,
                     _Float16* __restrict__ f1h,
                     _Float16* __restrict__ f2h) {
    const int gwave = (blockIdx.x * blockDim.x + threadIdx.x) >> 5;
    const int lane  = threadIdx.x & 31;

    const float* src;
    _Float16*    dst;
    int row;
    if (gwave < N_ROWS) { src = f1; dst = f1h; row = gwave; }
    else                { src = f2; dst = f2h; row = gwave - N_ROWS; }

    const v4f x = *(const v4f*)(src + (size_t)row * DIM + lane * 4);
    float s = x.x * x.x + x.y * x.y + x.z * x.z + x.w * x.w;
    s += __shfl_xor(s, 1,  32);
    s += __shfl_xor(s, 2,  32);
    s += __shfl_xor(s, 4,  32);
    s += __shfl_xor(s, 8,  32);
    s += __shfl_xor(s, 16, 32);
    const float inv = 1.0f / fmaxf(sqrtf(s), 1e-12f);  // F.normalize clamp

    v4h o;
    o.x = (_Float16)(x.x * inv);
    o.y = (_Float16)(x.y * inv);
    o.z = (_Float16)(x.z * inv);
    o.w = (_Float16)(x.w * inv);
    *(v4h*)(dst + (size_t)row * DIM + lane * 4) = o;
}

// ---------------------------------------------------------------------------
// Fragment loaders (layouts per CDNA5 ISA 7.12.2, wave32).
// ---------------------------------------------------------------------------
__device__ __forceinline__ void load_a_frags(AFrag* a, const _Float16* f1h,
                                             int row0, int lan, int half) {
    const _Float16* arow = f1h + (size_t)(row0 + lan) * DIM;
    #pragma unroll
    for (int kk = 0; kk < 4; ++kk) {
        a[kk].h[0] = *(const v8h*)(arow + kk * 32 + half * 8);       // K e<8
        a[kk].h[1] = *(const v8h*)(arow + kk * 32 + 16 + half * 8);  // K e>=8
    }
}

__device__ __forceinline__ void load_b_frags(AFrag* b, const _Float16* f2h,
                                             int j0, int lan, int half) {
    const _Float16* brow = f2h + (size_t)(j0 + lan) * DIM + half * 8;
    #pragma unroll
    for (int kk = 0; kk < 4; ++kk) {
        b[kk].h[0] = *(const v8h*)(brow + kk * 32);
        b[kk].h[1] = *(const v8h*)(brow + kk * 32 + 16);
    }
}

__device__ __forceinline__ v8f wmma_tile(const AFrag* a, const AFrag* b) {
    v8f acc = {};
    #pragma unroll
    for (int kk = 0; kk < 4; ++kk) {
        acc = __builtin_amdgcn_wmma_f32_16x16x32_f16(
            false, a[kk].v, false, b[kk].v, (short)0, acc, false, false);
    }
    return acc;
}

__device__ __forceinline__ float half_reduce16(float s) {
    // sum across the 16 lanes of a half-wave (xor masks 1..8 stay in-half)
    s += __shfl_xor(s, 1, 32);
    s += __shfl_xor(s, 2, 32);
    s += __shfl_xor(s, 4, 32);
    s += __shfl_xor(s, 8, 32);
    return s;
}

// ---------------------------------------------------------------------------
// Kernel 2: partial softmax denominators.  Grid (512 m-tiles, 8 j-chunks),
// one wave32 per block -> 4096 independent waves.  B tiles are double-
// buffered in VGPRs: prefetch tile j+1 while tile j's WMMAs + exps run.
// ---------------------------------------------------------------------------
__global__ void __launch_bounds__(32)
nce_zpartial_kernel(const _Float16* __restrict__ f1h,
                    const _Float16* __restrict__ f2h,
                    float* __restrict__ zp) {           // [CHUNKS][N_ROWS]
    const int lane = threadIdx.x;
    const int lan  = lane & 15;
    const int half = lane >> 4;
    const int row0 = blockIdx.x * 16;
    const int jbeg = blockIdx.y * JC;
    const int jend = jbeg + JC;

    AFrag a[4];
    load_a_frags(a, f1h, row0, lan, half);

    float z[8];
    #pragma unroll
    for (int r = 0; r < 8; ++r) z[r] = 0.0f;

    AFrag b0[4], b1[4];
    load_b_frags(b0, f2h, jbeg, lan, half);

    for (int j0 = jbeg; j0 < jend; j0 += 32) {
        // prefetch odd tile, compute even tile
        load_b_frags(b1, f2h, j0 + 16, lan, half);
        v8f acc = wmma_tile(a, b0);
        #pragma unroll
        for (int r = 0; r < 8; ++r) z[r] += __expf(acc[r]);   // TEMPERATURE=1

        // prefetch next even tile (uniform-guarded wrap), compute odd tile
        const int jn = (j0 + 32 < jend) ? (j0 + 32) : jbeg;
        load_b_frags(b0, f2h, jn, lan, half);
        acc = wmma_tile(a, b1);
        #pragma unroll
        for (int r = 0; r < 8; ++r) z[r] += __expf(acc[r]);
    }

    // lanes 0 and 16 each own rows row0+8*half .. +7 after the reduction
    #pragma unroll
    for (int r = 0; r < 8; ++r) {
        const float s = half_reduce16(z[r]);
        if (lan == 0) zp[(size_t)blockIdx.y * N_ROWS + row0 + half * 8 + r] = s;
    }
}

// ---------------------------------------------------------------------------
// Kernel 3: invZ[row] = 1 / sum_ch zp[ch][row]   (fixed chunk order)
// ---------------------------------------------------------------------------
__global__ void __launch_bounds__(256)
nce_zreduce_kernel(const float* __restrict__ zp, float* __restrict__ invZ) {
    const int row = blockIdx.x * 256 + threadIdx.x;
    float s = 0.0f;
    #pragma unroll
    for (int c = 0; c < CHUNKS; ++c) s += zp[(size_t)c * N_ROWS + row];
    invZ[row] = 1.0f / s;
}

// ---------------------------------------------------------------------------
// Kernel 4: masked-log loss partials.  Grid (512 m-tiles, 8 j-chunks).
// Double-buffered B operands; only ONE log per element.
// ---------------------------------------------------------------------------
__global__ void __launch_bounds__(32)
nce_loss_kernel(const _Float16* __restrict__ f1h,
                const _Float16* __restrict__ f2h,
                const float*    __restrict__ targets,
                const float*    __restrict__ invZ,
                float* __restrict__ SP,                 // [CHUNKS][N_ROWS]
                float* __restrict__ SN,
                float* __restrict__ CP) {
    const int lane = threadIdx.x;
    const int lan  = lane & 15;
    const int half = lane >> 4;
    const int row0 = blockIdx.x * 16;
    const int jbeg = blockIdx.y * JC;
    const int jend = jbeg + JC;

    AFrag a[4];
    load_a_frags(a, f1h, row0, lan, half);

    float ti[8], iz[8];
    #pragma unroll
    for (int r = 0; r < 8; ++r) {
        ti[r] = targets[row0 + half * 8 + r];
        iz[r] = invZ[row0 + half * 8 + r];
    }

    float sp[8], sn[8], cp[8];
    #pragma unroll
    for (int r = 0; r < 8; ++r) { sp[r] = 0.0f; sn[r] = 0.0f; cp[r] = 0.0f; }

    AFrag b0[4], b1[4];
    load_b_frags(b0, f2h, jbeg, lan, half);

    for (int j0 = jbeg; j0 < jend; j0 += 32) {
        // ---- even tile ----
        load_b_frags(b1, f2h, j0 + 16, lan, half);
        float tj = targets[j0 + lan];          // column index n == lan
        v8f acc = wmma_tile(a, b0);
        #pragma unroll
        for (int r = 0; r < 8; ++r) {
            const float p  = __expf(acc[r]) * iz[r];
            const float t  = ti[r];
            const bool  isneg = (fabsf(t - tj) > 0.1f) || ((t > 0.0f) != (tj > 0.0f));
            const float m  = isneg ? 1.0f : 0.0f;
            const float q  = isneg ? (1.0f - p) : p;
            const float val = -__logf(q + 1e-10f);     // single log per element
            sp[r] += (1.0f - m) * val;
            sn[r] += m * val;
            cp[r] += (1.0f - m);
        }

        // ---- odd tile ----
        const int jn = (j0 + 32 < jend) ? (j0 + 32) : jbeg;
        load_b_frags(b0, f2h, jn, lan, half);
        tj = targets[j0 + 16 + lan];
        acc = wmma_tile(a, b1);
        #pragma unroll
        for (int r = 0; r < 8; ++r) {
            const float p  = __expf(acc[r]) * iz[r];
            const float t  = ti[r];
            const bool  isneg = (fabsf(t - tj) > 0.1f) || ((t > 0.0f) != (tj > 0.0f));
            const float m  = isneg ? 1.0f : 0.0f;
            const float q  = isneg ? (1.0f - p) : p;
            const float val = -__logf(q + 1e-10f);
            sp[r] += (1.0f - m) * val;
            sn[r] += m * val;
            cp[r] += (1.0f - m);
        }
    }

    #pragma unroll
    for (int r = 0; r < 8; ++r) {
        const float rsp = half_reduce16(sp[r]);
        const float rsn = half_reduce16(sn[r]);
        const float rcp = half_reduce16(cp[r]);
        if (lan == 0) {
            const size_t idx = (size_t)blockIdx.y * N_ROWS + row0 + half * 8 + r;
            SP[idx] = rsp;
            SN[idx] = rsn;
            CP[idx] = rcp;
        }
    }
}

// ---------------------------------------------------------------------------
// Kernel 5: deterministic finalize.  Single block: per-row chunk sums in a
// fixed order, per-row combine, then a fixed LDS tree for the mean.
// ---------------------------------------------------------------------------
__global__ void __launch_bounds__(1024)
nce_finalize_kernel(const float* __restrict__ SP,
                    const float* __restrict__ SN,
                    const float* __restrict__ CP,
                    float* __restrict__ out) {
    __shared__ float smem[1024];
    float local = 0.0f;
    for (int row = threadIdx.x; row < N_ROWS; row += 1024) {
        float sp = 0.0f, sn = 0.0f, cp = 0.0f;
        #pragma unroll
        for (int c = 0; c < CHUNKS; ++c) {
            const size_t idx = (size_t)c * N_ROWS + row;
            sp += SP[idx];
            sn += SN[idx];
            cp += CP[idx];
        }
        local += sp / cp + sn / ((float)N_ROWS - cp);
    }
    smem[threadIdx.x] = local;
    __syncthreads();
    for (int off = 512; off > 0; off >>= 1) {
        if (threadIdx.x < (unsigned)off) smem[threadIdx.x] += smem[threadIdx.x + off];
        __syncthreads();
    }
    if (threadIdx.x == 0) out[0] = smem[0] * (1.0f / (float)N_ROWS);
}

// ---------------------------------------------------------------------------
extern "C" void kernel_launch(void* const* d_in, const int* in_sizes, int n_in,
                              void* d_out, int out_size, void* d_ws, size_t ws_size,
                              hipStream_t stream) {
    (void)in_sizes; (void)n_in; (void)out_size; (void)ws_size;
    const float* f1 = (const float*)d_in[0];
    const float* f2 = (const float*)d_in[1];
    const float* tg = (const float*)d_in[2];

    // workspace layout
    _Float16* f1h  = (_Float16*)d_ws;                         // 2 MB
    _Float16* f2h  = f1h + (size_t)N_ROWS * DIM;              // 2 MB
    float*    zp   = (float*)(f2h + (size_t)N_ROWS * DIM);    // 256 KB
    float*    invZ = zp   + (size_t)CHUNKS * N_ROWS;          // 32 KB
    float*    SP   = invZ + N_ROWS;                           // 256 KB
    float*    SN   = SP   + (size_t)CHUNKS * N_ROWS;          // 256 KB
    float*    CP   = SN   + (size_t)CHUNKS * N_ROWS;          // 256 KB

    // 2*8192 rows, one wave per row, 8 waves per 256-thread block
    nce_normalize_kernel<<<(2 * N_ROWS) / 8, 256, 0, stream>>>(f1, f2, f1h, f2h);

    // 512 m-tiles x 8 column chunks = 4096 independent wave32 blocks
    dim3 grid2(N_ROWS / 16, CHUNKS);
    nce_zpartial_kernel<<<grid2, 32, 0, stream>>>(f1h, f2h, zp);

    nce_zreduce_kernel<<<N_ROWS / 256, 256, 0, stream>>>(zp, invZ);

    nce_loss_kernel<<<grid2, 32, 0, stream>>>(f1h, f2h, tg, invZ, SP, SN, CP);

    nce_finalize_kernel<<<1, 1024, 0, stream>>>(SP, SN, CP, (float*)d_out);
}
